// XSA4Attention_79259326480693
// MI455X (gfx1250) — compile-verified, matrix operand order checked by reading, not run on hardware
//
#include <hip/hip_runtime.h>
#include <hip/hip_bf16.h>

// ---------------------------------------------------------------------------
// Causal MHA block: qkv proj (+RoPE) -> flash attention -> out proj.
// B=4, T=2048, D=1024, H=16, hd=64, rope_dims=32.
// Matmuls: v_wmma_f32_16x16x32_f16.  Operand staging: TDM tensor_load_to_lds
// double-buffered in LDS with s_wait_tensorcnt pipelining (CDNA5 path).
// ---------------------------------------------------------------------------

#define BB 4
#define TT 2048
#define DD 1024
#define HH 16
#define HD 64
#define D3 (3 * DD)

typedef __attribute__((ext_vector_type(16))) _Float16 v16h;
typedef __attribute__((ext_vector_type(8)))  float    v8f;
typedef __attribute__((ext_vector_type(4)))  unsigned int v4u;
typedef __attribute__((ext_vector_type(8)))  int      v8i;
typedef __attribute__((ext_vector_type(4)))  int      v4i;

#define WMMA_F16(A, B, C) \
  __builtin_amdgcn_wmma_f32_16x16x32_f16(false, (A), false, (B), (short)0, (C), false, false)

// --- TDM: 2D tile (f16 elements) global -> LDS ------------------------------
// D# per CDNA5 ISA ch.8: group0 = {count, lds_addr, global_addr, type=2};
// group1 = {data_size=1(2B), tensor dims, tile dims, dim0 stride}; 2D => g2=g3=0.
// This toolchain exposes the 6-arg builtin (extra int32x8 group before cpol).
__device__ __forceinline__ void tdm_load_2d_f16(const _Float16* lds_dst,
                                                const _Float16* gsrc,
                                                uint32_t td0, uint32_t td1,
                                                uint32_t tile0, uint32_t tile1,
                                                uint32_t stride0) {
  uint32_t lds_off = (uint32_t)(uintptr_t)lds_dst;   // LDS offset: low 32 bits of the pointer
  uint64_t ga = (uint64_t)(uintptr_t)gsrc;
  v4u g0 = { 1u,                                     // count = 1 valid descriptor
             lds_off,
             (uint32_t)ga,
             (uint32_t)(ga >> 32) | (2u << 30) };    // type = 2 ("image")
  v8i g1 = { (int)0x00010000u,                       // data_size = 2 bytes
             (int)((td0 & 0xFFFFu) << 16),           // tensor_dim0[15:0]
             (int)((td0 >> 16) | ((td1 & 0xFFFFu) << 16)),
             (int)((td1 >> 16) | (tile0 << 16)),     // tile_dim0
             (int)tile1,                             // tile_dim1 (tile_dim2 = 0)
             (int)stride0,                           // tensor_dim0_stride[31:0]
             0, 0 };
  v4i z4 = { 0, 0, 0, 0 };
  v8i z8 = { 0, 0, 0, 0, 0, 0, 0, 0 };
  __builtin_amdgcn_tensor_load_to_lds(g0, g1, z4, z4, z8, 0);
}

// --- fragment loaders (ISA 7.12.2 layouts) ----------------------------------
__device__ __forceinline__ v16h load_frag_rm_f16(const _Float16* __restrict__ p,
                                                 int ld, int lane) {
  const int r  = lane & 15;
  const int kb = (lane >> 4) << 3;
  const _Float16* b = p + (size_t)r * ld + kb;
  v16h f;
#pragma unroll
  for (int i = 0; i < 8; ++i) { f[i] = b[i]; f[i + 8] = b[i + 16]; }
  return f;
}

__device__ __forceinline__ v16h load_frag_rm_f32(const float* __restrict__ p,
                                                 int ld, int lane) {
  const int r  = lane & 15;
  const int kb = (lane >> 4) << 3;
  const float* b = p + (size_t)r * ld + kb;
  v16h f;
#pragma unroll
  for (int i = 0; i < 8; ++i) {
    f[i]     = (_Float16)b[i];
    f[i + 8] = (_Float16)b[i + 16];
  }
  return f;
}

// B fragment from row-major [K, N] source (K strided): used for P x V.
__device__ __forceinline__ v16h load_frag_kn_f16(const _Float16* __restrict__ p,
                                                 int ld, int lane) {
  const int c  = lane & 15;
  const int kb = (lane >> 4) << 3;
  v16h f;
#pragma unroll
  for (int i = 0; i < 8; ++i) {
    f[i]     = p[(size_t)(kb + i) * ld + c];
    f[i + 8] = p[(size_t)(kb + 16 + i) * ld + c];
  }
  return f;
}

__device__ __forceinline__ void store_tile_f16(_Float16* __restrict__ p, int ld,
                                               v8f c, int lane) {
  const int col = lane & 15;
  const int rb  = (lane >> 4) << 3;
#pragma unroll
  for (int r = 0; r < 8; ++r) p[(size_t)(rb + r) * ld + col] = (_Float16)c[r];
}

__device__ __forceinline__ void store_tile_f32(float* __restrict__ p, int ld,
                                               v8f c, int lane) {
  const int col = lane & 15;
  const int rb  = (lane >> 4) << 3;
#pragma unroll
  for (int r = 0; r < 8; ++r) p[(size_t)(rb + r) * ld + col] = c[r];
}

// --- kernel 0: f32 -> f16 conversion ----------------------------------------
__global__ void cvt_f32_to_f16_kernel(const float* __restrict__ s,
                                      _Float16* __restrict__ d, int n) {
  int i = blockIdx.x * blockDim.x + threadIdx.x;
  if (i < n) d[i] = (_Float16)s[i];
}

// --- kernel 1: qkv = x @ Wqkv^T ---------------------------------------------
// Block = 8 waves sharing one 64-col W window; W k-tiles (64x32 f16 = 4KB)
// TDM-staged into LDS, double buffered; wave = 16x64 output tile.
__global__ void __launch_bounds__(256)
qkv_gemm_kernel(const float* __restrict__ X, const _Float16* __restrict__ W,
                _Float16* __restrict__ C) {
  __shared__ __align__(16) _Float16 bstage[2][64 * 32];   // 8 KB
  const int lane = threadIdx.x & 31;
  const int wid  = threadIdx.x >> 5;
  const int m0   = (blockIdx.y * 8 + wid) << 4;           // over B*T = 8192
  const int n0   = blockIdx.x << 6;                       // over 3D  = 3072
  const _Float16* wbase = W + (size_t)n0 * DD;
  const float*    xp    = X + (size_t)m0 * DD;

  if (wid == 0)  // prologue: stage k-tile 0
    tdm_load_2d_f16(&bstage[0][0], wbase, DD, 1u << 20, 32, 64, DD);

  v8f acc[4] = {};
  const int ksteps = DD / 32;                             // 32
  for (int j = 0; j < ksteps; ++j) {
    // A fragment load overlaps the tensor wait
    v16h a = load_frag_rm_f32(xp + j * 32, DD, lane);
    if (wid == 0) {
      if (j + 1 < ksteps) {
        tdm_load_2d_f16(&bstage[(j + 1) & 1][0], wbase + (j + 1) * 32,
                        DD, 1u << 20, 32, 64, DD);
        __builtin_amdgcn_s_wait_tensorcnt(1);             // tile j landed
      } else {
        __builtin_amdgcn_s_wait_tensorcnt(0);
      }
    }
    __syncthreads();                                      // publish tile j
    const _Float16* bs = &bstage[j & 1][0];
#pragma unroll
    for (int i = 0; i < 4; ++i) {
      v16h b = load_frag_rm_f16(bs + (i * 16) * 32, 32, lane);
      acc[i] = WMMA_F16(a, b, acc[i]);
    }
    __syncthreads();                                      // retire tile j
  }
#pragma unroll
  for (int i = 0; i < 4; ++i)
    store_tile_f16(C + (size_t)m0 * D3 + n0 + i * 16, D3, acc[i], lane);
}

// --- kernel 2: RoPE + transpose [B,T,3,H,hd] -> Q/K/V [B,H,T,hd] ------------
__global__ void rope_transpose_kernel(const _Float16* __restrict__ QKV,
                                      _Float16* __restrict__ Q,
                                      _Float16* __restrict__ K2,
                                      _Float16* __restrict__ V2) {
  const int per = BB * HH * TT * HD;  // 8388608
  int tid = blockIdx.x * blockDim.x + threadIdx.x;
  if (tid >= 3 * per) return;
  int comp = tid / per;
  int r    = tid - comp * per;        // ((b*H+h)*T + t)*hd + d
  int d = r & 63;
  int t = (r >> 6) & (TT - 1);
  int h = (r >> 17) & (HH - 1);
  int b = r >> 21;
  const _Float16* src = QKV + (size_t)(b * TT + t) * D3 + comp * DD + h * HD;
  float val;
  if (comp < 2 && d < 32) {           // rope_dims = 32, half = 16
    int i = d & 15;
    float inv = __expf(-(float)i * (9.210340371976184f / 16.0f)); // 10000^(-i/16)
    float ang = (float)t * inv;
    float c = __cosf(ang), sn = __sinf(ang);
    float x1 = (float)src[i];
    float x2 = (float)src[i + 16];
    val = (d < 16) ? (x1 * c - x2 * sn) : (x2 * c + x1 * sn);
  } else {
    val = (float)src[d];
  }
  _Float16* dst = (comp == 0) ? Q : (comp == 1) ? K2 : V2;
  dst[r] = (_Float16)val;
}

// --- kernel 3: flash attention ----------------------------------------------
// Block = 4 waves = 4 consecutive q-tiles of the SAME (b,h).  K/V 32-key
// blocks (32x64 f16 = 4KB each) TDM-staged into LDS once per block, double
// buffered; per-wave causal early-out inside uniform barriers.
__global__ void __launch_bounds__(128)
flash_attn_kernel(const _Float16* __restrict__ Q, const _Float16* __restrict__ K,
                  const _Float16* __restrict__ V, _Float16* __restrict__ O) {
  __shared__ __align__(16) _Float16 kbuf[2][32 * 64];   // 8 KB
  __shared__ __align__(16) _Float16 vbuf[2][32 * 64];   // 8 KB
  __shared__ __align__(16) _Float16 pbuf[4][16 * 40];   // 5 KB: per-wave P tile
  const int lane = threadIdx.x & 31;
  const int wid  = threadIdx.x >> 5;
  const int qt   = blockIdx.x * 4 + wid;    // 8192 q-tiles; 128 per (b,h)
  const int bh   = qt >> 7;
  const int q0   = (qt & 127) << 4;
  const int b    = bh >> 4;
  const int h    = bh & 15;
  const int rb   = (lane >> 4) << 3;
  const int cl   = lane & 15;
  const float scale = 0.125f;               // 1/sqrt(64)
  const _Float16* kvbase = K + (size_t)bh * TT * HD;
  const _Float16* vvbase = V + (size_t)bh * TT * HD;

  const size_t qoff = (size_t)(bh * TT + q0) * HD;
  v16h qa0 = load_frag_rm_f16(Q + qoff, HD, lane);
  v16h qa1 = load_frag_rm_f16(Q + qoff + 32, HD, lane);

  v8f o[4] = {};
  float m[8], l[8];
#pragma unroll
  for (int r = 0; r < 8; ++r) { m[r] = -1e30f; l[r] = 0.0f; }

  const int nb_own = (q0 + 47) >> 5;
  const int q0max  = (((blockIdx.x * 4 + 3) & 127) << 4);
  const int nb_max = (q0max + 47) >> 5;     // uniform loop bound for the block

  if (wid == 0) {                           // prologue: stage key-block 0
    tdm_load_2d_f16(&kbuf[0][0], kvbase, HD, 1u << 20, HD, 32, HD);
    tdm_load_2d_f16(&vbuf[0][0], vvbase, HD, 1u << 20, HD, 32, HD);
  }

  for (int j = 0; j < nb_max; ++j) {
    const int n0 = j << 5;
    if (wid == 0) {
      if (j + 1 < nb_max) {
        const size_t noff = (size_t)(n0 + 32) * HD;
        tdm_load_2d_f16(&kbuf[(j + 1) & 1][0], kvbase + noff, HD, 1u << 20, HD, 32, HD);
        tdm_load_2d_f16(&vbuf[(j + 1) & 1][0], vvbase + noff, HD, 1u << 20, HD, 32, HD);
        __builtin_amdgcn_s_wait_tensorcnt(2);   // pair j landed
      } else {
        __builtin_amdgcn_s_wait_tensorcnt(0);
      }
    }
    __syncthreads();                            // publish K/V block j

    if (j < nb_own) {
      const _Float16* kt = &kbuf[j & 1][0];     // [32 keys][64 d]
      const _Float16* vt = &vbuf[j & 1][0];
      v8f s[2];
#pragma unroll
      for (int t2 = 0; t2 < 2; ++t2) {
        v16h b0 = load_frag_rm_f16(kt + (16 * t2) * HD, HD, lane);
        v16h b1 = load_frag_rm_f16(kt + (16 * t2) * HD + 32, HD, lane);
        v8f sv = {};
        sv = WMMA_F16(qa0, b0, sv);
        sv = WMMA_F16(qa1, b1, sv);
        s[t2] = sv;
      }
      const bool need_mask = (n0 + 31) > q0;
      float alpha[8];
#pragma unroll
      for (int r = 0; r < 8; ++r) {
        float e0 = s[0][r] * scale;
        float e1 = s[1][r] * scale;
        if (need_mask) {
          int row = q0 + rb + r;
          if (n0 + cl > row)      e0 = -1e30f;
          if (n0 + 16 + cl > row) e1 = -1e30f;
        }
        float v = fmaxf(e0, e1);                // row reduce (16-lane half)
        v = fmaxf(v, __shfl_xor(v, 1, 32));
        v = fmaxf(v, __shfl_xor(v, 2, 32));
        v = fmaxf(v, __shfl_xor(v, 4, 32));
        v = fmaxf(v, __shfl_xor(v, 8, 32));
        float nm = fmaxf(m[r], v);
        float a  = __expf(m[r] - nm);
        m[r] = nm;
        float p0 = __expf(e0 - nm);
        float p1 = __expf(e1 - nm);
        float rs = p0 + p1;
        rs += __shfl_xor(rs, 1, 32);
        rs += __shfl_xor(rs, 2, 32);
        rs += __shfl_xor(rs, 4, 32);
        rs += __shfl_xor(rs, 8, 32);
        l[r] = l[r] * a + rs;
        alpha[r] = a;
        pbuf[wid][(rb + r) * 40 + cl]      = (_Float16)p0;  // C->A transpose
        pbuf[wid][(rb + r) * 40 + 16 + cl] = (_Float16)p1;
      }
#pragma unroll
      for (int i = 0; i < 4; ++i)
#pragma unroll
        for (int r = 0; r < 8; ++r) o[i][r] *= alpha[r];

      v16h pa = load_frag_rm_f16(&pbuf[wid][0], 40, lane);
#pragma unroll
      for (int i = 0; i < 4; ++i) {
        v16h vb = load_frag_kn_f16(vt + i * 16, HD, lane);
        o[i] = WMMA_F16(pa, vb, o[i]);
      }
    }
    __syncthreads();                            // retire K/V block j
  }

#pragma unroll
  for (int r = 0; r < 8; ++r) {
    const float inv = 1.0f / l[r];
    const int row = q0 + rb + r;
    _Float16* op = O + (size_t)(b * TT + row) * DD + h * HD + cl;
#pragma unroll
    for (int i = 0; i < 4; ++i) op[i * 16] = (_Float16)(o[i][r] * inv);
  }
}

// --- kernel 4: out = attn @ Wout^T  (f16 x f16 -> f32) ----------------------
__global__ void __launch_bounds__(256)
out_gemm_kernel(const _Float16* __restrict__ A, const _Float16* __restrict__ W,
                float* __restrict__ C) {
  __shared__ __align__(16) _Float16 bstage[2][64 * 32];   // 8 KB
  const int lane = threadIdx.x & 31;
  const int wid  = threadIdx.x >> 5;
  const int m0   = (blockIdx.y * 8 + wid) << 4;
  const int n0   = blockIdx.x << 6;
  const _Float16* wbase = W + (size_t)n0 * DD;
  const _Float16* ap    = A + (size_t)m0 * DD;

  if (wid == 0)
    tdm_load_2d_f16(&bstage[0][0], wbase, DD, 1u << 20, 32, 64, DD);

  v8f acc[4] = {};
  const int ksteps = DD / 32;
  for (int j = 0; j < ksteps; ++j) {
    v16h a = load_frag_rm_f16(ap + j * 32, DD, lane);
    if (wid == 0) {
      if (j + 1 < ksteps) {
        tdm_load_2d_f16(&bstage[(j + 1) & 1][0], wbase + (j + 1) * 32,
                        DD, 1u << 20, 32, 64, DD);
        __builtin_amdgcn_s_wait_tensorcnt(1);
      } else {
        __builtin_amdgcn_s_wait_tensorcnt(0);
      }
    }
    __syncthreads();
    const _Float16* bs = &bstage[j & 1][0];
#pragma unroll
    for (int i = 0; i < 4; ++i) {
      v16h b = load_frag_rm_f16(bs + (i * 16) * 32, 32, lane);
      acc[i] = WMMA_F16(a, b, acc[i]);
    }
    __syncthreads();
  }
#pragma unroll
  for (int i = 0; i < 4; ++i)
    store_tile_f32(C + (size_t)m0 * DD + n0 + i * 16, DD, acc[i], lane);
}

// ---------------------------------------------------------------------------
extern "C" void kernel_launch(void* const* d_in, const int* in_sizes, int n_in,
                              void* d_out, int out_size, void* d_ws, size_t ws_size,
                              hipStream_t stream) {
  const float* x    = (const float*)d_in[0];   // [B,T,D]
  const float* Wqkv = (const float*)d_in[1];   // [3D,D]
  const float* Wout = (const float*)d_in[2];   // [D,D]
  float* out = (float*)d_out;                  // [B,T,D]

  // f16 workspace layout (126 MB total)
  _Float16* wqkvh = (_Float16*)d_ws;                  // 3D*D
  _Float16* wouth = wqkvh + (size_t)D3 * DD;          // D*D
  _Float16* qkvh  = wouth + (size_t)DD * DD;          // B*T*3D
  _Float16* Qh    = qkvh  + (size_t)BB * TT * D3;     // B*H*T*hd
  _Float16* Kh    = Qh    + (size_t)BB * HH * TT * HD;
  _Float16* Vh    = Kh    + (size_t)BB * HH * TT * HD;
  _Float16* attnh = Vh    + (size_t)BB * HH * TT * HD;

  cvt_f32_to_f16_kernel<<<(D3 * DD + 255) / 256, 256, 0, stream>>>(Wqkv, wqkvh, D3 * DD);
  cvt_f32_to_f16_kernel<<<(DD * DD + 255) / 256, 256, 0, stream>>>(Wout, wouth, DD * DD);
  qkv_gemm_kernel<<<dim3(D3 / 64, (BB * TT) / 16 / 8), 256, 0, stream>>>(x, wqkvh, qkvh);
  {
    int n = 3 * BB * HH * TT * HD;
    rope_transpose_kernel<<<(n + 255) / 256, 256, 0, stream>>>(qkvh, Qh, Kh, Vh);
  }
  flash_attn_kernel<<<(BB * HH * (TT / 16)) / 4, 128, 0, stream>>>(Qh, Kh, Vh, attnh);
  out_gemm_kernel<<<dim3(DD / 64, (BB * TT) / 16 / 8), 256, 0, stream>>>(attnh, wouth, out);
}
